// DNL_4320737099853
// MI455X (gfx1250) — matrix operand order repaired and checked by reading, not compiled, not run on hardware
//
#include <hip/hip_runtime.h>

// ---------------------------------------------------------------------------
// Types for CDNA5 WMMA (gfx1250, wave32)
// ---------------------------------------------------------------------------
typedef unsigned short u16;
typedef __attribute__((ext_vector_type(16))) __bf16 v16bf;
typedef __attribute__((ext_vector_type(8)))  __bf16 v8bf;
typedef __attribute__((ext_vector_type(8)))  float  v8f;

static constexpr int  BATCH = 2;
static constexpr int  CH    = 512;
static constexpr int  HS    = 48;
static constexpr int  HWS   = 48 * 48;    // 2304
static constexpr int  H2    = 96;
static constexpr int  HW2   = 96 * 96;    // 9216
static constexpr int  C8    = 64;
static constexpr int  NCLS  = 19;
static constexpr int  NPAD  = 64;         // class dim padded for K=64 GEMM tile
static constexpr float EPSV = 1e-5f;

__device__ __forceinline__ u16 f2bf(float f) {
  union { float f; unsigned u; } cv; cv.f = f;
  unsigned r = 0x7FFFu + ((cv.u >> 16) & 1u);   // round-to-nearest-even
  return (u16)((cv.u + r) >> 16);
}

// Assemble a 16-element bf16 fragment from two contiguous 16B LDS chunks.
__device__ __forceinline__ v16bf ldfrag(const u16* p0, const u16* p1) {
  v8bf lo = *(const v8bf*)p0;
  v8bf hi = *(const v8bf*)p1;
  return __builtin_shufflevector(lo, hi, 0,1,2,3,4,5,6,7,8,9,10,11,12,13,14,15);
}

// CDNA5 async global->LDS copy (16B per lane), tracked with ASYNCcnt.
// Generic pointers to __shared__ carry the LDS byte offset in the low 32 bits
// (ISA 10.2: LDS aperture address truncation), so we pass that as the LDS addr.
__device__ __forceinline__ void async_ld_b128(const u16* lds_ptr, const u16* gptr) {
  unsigned lds_off = (unsigned)(unsigned long long)lds_ptr;
  asm volatile("global_load_async_to_lds_b128 %0, %1, off"
               :: "v"(lds_off), "v"(gptr) : "memory");
}
__device__ __forceinline__ void wait_async() {
  asm volatile("s_wait_asynccnt 0x0" ::: "memory");
}

#define WMMA_BF16(A_, B_, C_) \
  __builtin_amdgcn_wmma_f32_16x16x32_bf16(false, (A_), false, (B_), (short)0, (C_), false, false)

// ---------------------------------------------------------------------------
// Batched bf16 GEMM: C[M,N](f32) = A[M,K](bf16,row) * B[K,N](bf16,row)
// Block tile 64x128, K tile 64, 256 threads = 8 waves in a 2(M) x 4(N) grid;
// each wave computes a 32x32 output = 2x2 WMMA tiles -> 8 WMMAs per K tile.
// A tile staged via global_load_async_to_lds_b128 (straight copy).
// B tile transposed in registers: 4x ushort4 loads -> 4x ds_store_b64.
// ---------------------------------------------------------------------------
#define BLK_M 64
#define BLK_N 128
#define BLK_K 64

__global__ void __launch_bounds__(256)
gemm_bf16_k(const u16* __restrict__ A, const u16* __restrict__ B,
            float* __restrict__ C, int M, int N, int K,
            int lda, int ldb, int ldc,
            long long sA, long long sB, long long sC)
{
  __shared__ __align__(16) u16 At[BLK_M][BLK_K];
  __shared__ __align__(16) u16 Bt[BLK_N][BLK_K];
  const int zb = blockIdx.z;
  A += (size_t)zb * sA; B += (size_t)zb * sB; C += (size_t)zb * sC;
  const int n0 = blockIdx.x * BLK_N;
  const int m0 = blockIdx.y * BLK_M;
  const int t    = threadIdx.x;
  const int lane = t & 31, wave = t >> 5;
  const int wm = (wave & 1) << 5, wn = (wave >> 1) << 5;
  const int lg = lane >> 4, ln = lane & 15;
  v8f acc00 = {}, acc01 = {}, acc10 = {}, acc11 = {};
  for (int k0 = 0; k0 < K; k0 += BLK_K) {
    { // A tile 64x64 halves: two async B128 per thread
      int m = t >> 2, k = (t & 3) << 4;
      const u16* ga = &A[(size_t)(m0 + m) * lda + (k0 + k)];
      async_ld_b128(&At[m][k], ga);
      async_ld_b128(&At[m][k + 8], ga + 8);
    }
    // B tile 64(K)x128(N) -> Bt[n][k], register transpose of 4x4 blocks
    #pragma unroll
    for (int u = 0; u < 2; ++u) {
      int uu = t + u * 256;               // 0..511
      int n  = (uu & 31) << 2;            // 0..124
      int k  = (uu >> 5) << 2;            // 0..60
      const u16* bp = &B[(size_t)(k0 + k) * ldb + n0 + n];
      ushort4 r0 = *(const ushort4*)(bp);
      ushort4 r1 = *(const ushort4*)(bp + ldb);
      ushort4 r2 = *(const ushort4*)(bp + 2 * (size_t)ldb);
      ushort4 r3 = *(const ushort4*)(bp + 3 * (size_t)ldb);
      *(ushort4*)&Bt[n + 0][k] = make_ushort4(r0.x, r1.x, r2.x, r3.x);
      *(ushort4*)&Bt[n + 1][k] = make_ushort4(r0.y, r1.y, r2.y, r3.y);
      *(ushort4*)&Bt[n + 2][k] = make_ushort4(r0.z, r1.z, r2.z, r3.z);
      *(ushort4*)&Bt[n + 3][k] = make_ushort4(r0.w, r1.w, r2.w, r3.w);
    }
    wait_async();
    __syncthreads();
    #pragma unroll
    for (int kk = 0; kk < BLK_K; kk += 32) {
      v16bf a0 = ldfrag(&At[wm + ln][kk + 8 * lg],      &At[wm + ln][kk + 16 + 8 * lg]);
      v16bf a1 = ldfrag(&At[wm + 16 + ln][kk + 8 * lg], &At[wm + 16 + ln][kk + 16 + 8 * lg]);
      v16bf b0 = ldfrag(&Bt[wn + ln][kk + 8 * lg],      &Bt[wn + ln][kk + 16 + 8 * lg]);
      v16bf b1 = ldfrag(&Bt[wn + 16 + ln][kk + 8 * lg], &Bt[wn + 16 + ln][kk + 16 + 8 * lg]);
      acc00 = WMMA_BF16(a0, b0, acc00);
      acc01 = WMMA_BF16(a0, b1, acc01);
      acc10 = WMMA_BF16(a1, b0, acc10);
      acc11 = WMMA_BF16(a1, b1, acc11);
    }
    __syncthreads();
  }
  float* Cp0 = C + (size_t)(m0 + wm + 8 * lg) * ldc + (n0 + wn + ln);
  float* Cp1 = Cp0 + (size_t)16 * ldc;
  #pragma unroll
  for (int r = 0; r < 8; ++r) {
    Cp0[(size_t)r * ldc]      = acc00[r];
    Cp0[(size_t)r * ldc + 16] = acc01[r];
    Cp1[(size_t)r * ldc]      = acc10[r];
    Cp1[(size_t)r * ldc + 16] = acc11[r];
  }
}

// ---------------------------------------------------------------------------
// Implicit-GEMM 3x3 conv, pad=1, stride=1. A = weights bf16 [Cout, Cin*9]
// (OIHW flat, async-staged); B tile built by im2col into LDS.
// N enumerates n = b*H*W + h*W + w. Out f32 [B, Cout, H*W].
// ---------------------------------------------------------------------------
__global__ void __launch_bounds__(256)
conv3x3_gemm_k(const u16* __restrict__ Wt, const u16* __restrict__ In,
               float* __restrict__ Out, int Cout, int Cin, int H, int W, int Bn)
{
  __shared__ __align__(16) u16 At[BLK_M][BLK_K];
  __shared__ __align__(16) u16 Bt[BLK_N][BLK_K];
  const int K   = Cin * 9;
  const int HWl = H * W;
  const int n0 = blockIdx.x * BLK_N;
  const int m0 = blockIdx.y * BLK_M;
  const int t    = threadIdx.x;
  const int lane = t & 31, wave = t >> 5;
  const int wm = (wave & 1) << 5, wn = (wave >> 1) << 5;
  const int lg = lane >> 4, ln = lane & 15;
  v8f acc00 = {}, acc01 = {}, acc10 = {}, acc11 = {};
  for (int k0 = 0; k0 < K; k0 += BLK_K) {
    { // weight tile via async copy
      int m = t >> 2, k = (t & 3) << 4;
      const u16* ga = &Wt[(size_t)(m0 + m) * K + (k0 + k)];
      async_ld_b128(&At[m][k], ga);
      async_ld_b128(&At[m][k + 8], ga + 8);
    }
    // im2col B tile: 128 n x 16 k-quads = 2048 units, 8 per thread
    #pragma unroll
    for (int u = 0; u < 8; ++u) {
      int uu = t + u * 256;               // 0..2047
      int nl = uu & 127, kg = uu >> 7;    // kg 0..15
      int n  = n0 + nl;
      int bb = n / HWl, hw = n % HWl;
      int hb = hw / W, wb = hw % W;
      u16 vals[4];
      #pragma unroll
      for (int j = 0; j < 4; ++j) {
        int kk = k0 + kg * 4 + j;
        int c = kk / 9, r9 = kk % 9;
        int h = hb + r9 / 3 - 1, w = wb + r9 % 3 - 1;
        u16 val = 0;
        if (h >= 0 && h < H && w >= 0 && w < W)
          val = In[((size_t)bb * Cin + c) * HWl + h * W + w];
        vals[j] = val;
      }
      *(ushort4*)&Bt[nl][kg * 4] = make_ushort4(vals[0], vals[1], vals[2], vals[3]);
    }
    wait_async();
    __syncthreads();
    #pragma unroll
    for (int kk = 0; kk < BLK_K; kk += 32) {
      v16bf a0 = ldfrag(&At[wm + ln][kk + 8 * lg],      &At[wm + ln][kk + 16 + 8 * lg]);
      v16bf a1 = ldfrag(&At[wm + 16 + ln][kk + 8 * lg], &At[wm + 16 + ln][kk + 16 + 8 * lg]);
      v16bf b0 = ldfrag(&Bt[wn + ln][kk + 8 * lg],      &Bt[wn + ln][kk + 16 + 8 * lg]);
      v16bf b1 = ldfrag(&Bt[wn + 16 + ln][kk + 8 * lg], &Bt[wn + 16 + ln][kk + 16 + 8 * lg]);
      acc00 = WMMA_BF16(a0, b0, acc00);
      acc01 = WMMA_BF16(a0, b1, acc01);
      acc10 = WMMA_BF16(a1, b0, acc10);
      acc11 = WMMA_BF16(a1, b1, acc11);
    }
    __syncthreads();
  }
  const int na = n0 + wn + ln;
  const int nb = na + 16;
  const int ba = na / HWl, ha = na % HWl;
  const int bbn = nb / HWl, hbn = nb % HWl;
  const int ob = m0 + wm + 8 * lg;
  #pragma unroll
  for (int r = 0; r < 8; ++r) {
    Out[((size_t)ba  * Cout + ob + r)      * HWl + ha]  = acc00[r];
    Out[((size_t)bbn * Cout + ob + r)      * HWl + hbn] = acc01[r];
    Out[((size_t)ba  * Cout + ob + 16 + r) * HWl + ha]  = acc10[r];
    Out[((size_t)bbn * Cout + ob + 16 + r) * HWl + hbn] = acc11[r];
  }
}

// ---------------------------------------------------------------------------
// Elementwise / reduction kernels (f32 VALU)
// ---------------------------------------------------------------------------
__global__ void cvt_bf16_k(const float* __restrict__ s, u16* __restrict__ d, long long n) {
  long long stride = (long long)gridDim.x * 256;
  for (long long i = blockIdx.x * 256LL + threadIdx.x; i < n; i += stride)
    d[i] = f2bf(s[i]);
}

// zero-padded row convert: d has rowsPad rows, first `rows` copied, rest 0
__global__ void cvt_pad_rows_k(const float* __restrict__ s, u16* __restrict__ d,
                               int rows, int rowsPad, int cols) {
  long long n = (long long)rowsPad * cols;
  long long stride = (long long)gridDim.x * 256;
  for (long long i = blockIdx.x * 256LL + threadIdx.x; i < n; i += stride) {
    int r = (int)(i / cols);
    d[i] = (r < rows) ? f2bf(s[i]) : (u16)0;
  }
}

// transpose+convert: src [R,Cc] f32 -> dst [Cc,R] bf16 (batched via z)
__global__ void tcvt_bf16_k(const float* __restrict__ s, u16* __restrict__ d,
                            int R, int Cc, long long sS, long long sD) {
  s += (size_t)blockIdx.z * sS; d += (size_t)blockIdx.z * sD;
  long long n = (long long)R * Cc;
  long long stride = (long long)gridDim.x * 256;
  for (long long i = blockIdx.x * 256LL + threadIdx.x; i < n; i += stride) {
    long long c = i / R, r = i % R;
    d[i] = f2bf(s[r * (long long)Cc + c]);
  }
}

// local_relation[b,n,p] = sigmoid(sum_c q[b,c,n] * k[b,c,shift_p(n)])
__global__ void local_rel_k(const float* __restrict__ q, const float* __restrict__ k,
                            float* __restrict__ lr) {
  long long n = (long long)BATCH * HWS * 9;
  long long stride = (long long)gridDim.x * 256;
  for (long long i = blockIdx.x * 256LL + threadIdx.x; i < n; i += stride) {
    int p = (int)(i % 9); long long bn = i / 9;
    int nn = (int)(bn % HWS); int b = (int)(bn / HWS);
    int h = nn / HS + p / 3 - 1, w = nn % HS + p % 3 - 1;
    float s = 0.f;
    if (h >= 0 && h < HS && w >= 0 && w < HS) {
      int m = h * HS + w;
      const float* qp = q + (size_t)b * C8 * HWS + nn;
      const float* kp = k + (size_t)b * C8 * HWS + m;
      #pragma unroll 4
      for (int c = 0; c < C8; ++c) s += qp[(size_t)c * HWS] * kp[(size_t)c * HWS];
    }
    lr[i] = 1.f / (1.f + __expf(-s));
  }
}

// per-channel biased mean/var over (B, HW)
__global__ void __launch_bounds__(256)
bn_stats_k(const float* __restrict__ x, float* __restrict__ mean,
           float* __restrict__ var, int Cch, long long HWn, int Bc) {
  __shared__ float s1[256], s2[256];
  int c = blockIdx.x, t = threadIdx.x;
  float a = 0.f, b = 0.f;
  for (int bb = 0; bb < Bc; ++bb) {
    const float* p = x + ((size_t)bb * Cch + c) * HWn;
    for (long long i = t; i < HWn; i += 256) { float v = p[i]; a += v; b += v * v; }
  }
  s1[t] = a; s2[t] = b; __syncthreads();
  for (int o = 128; o > 0; o >>= 1) {
    if (t < o) { s1[t] += s1[t + o]; s2[t] += s2[t + o]; }
    __syncthreads();
  }
  if (t == 0) {
    float n = (float)((double)Bc * (double)HWn);
    float m = s1[0] / n;
    mean[c] = m; var[c] = s2[0] / n - m * m;
  }
}

__global__ void bn_relu_bf16_k(const float* __restrict__ x, const float* __restrict__ mean,
                               const float* __restrict__ var, const float* __restrict__ g,
                               const float* __restrict__ bta, u16* __restrict__ out,
                               int Cch, long long HWn, long long total) {
  long long stride = (long long)gridDim.x * 256;
  for (long long i = blockIdx.x * 256LL + threadIdx.x; i < total; i += stride) {
    int c = (int)((i / HWn) % Cch);
    float v = (x[i] - mean[c]) * rsqrtf(var[c] + EPSV) * g[c] + bta[c];
    out[i] = f2bf(v > 0.f ? v : 0.f);
  }
}

__global__ void bn_relu_f32_k(const float* __restrict__ x, const float* __restrict__ mean,
                              const float* __restrict__ var, const float* __restrict__ g,
                              const float* __restrict__ bta, float* __restrict__ out,
                              int Cch, long long HWn, long long total) {
  long long stride = (long long)gridDim.x * 256;
  for (long long i = blockIdx.x * 256LL + threadIdx.x; i < total; i += stride) {
    int c = (int)((i / HWn) % Cch);
    float v = (x[i] - mean[c]) * rsqrtf(var[c] + EPSV) * g[c] + bta[c];
    out[i] = v > 0.f ? v : 0.f;
  }
}

// aux2[b,o,n] = tmp[b,o(pad64),n] + bias[o]
__global__ void aux2_fin_k(const float* __restrict__ tmp, const float* __restrict__ bias,
                           float* __restrict__ out) {
  long long n = (long long)BATCH * NCLS * HWS;
  long long stride = (long long)gridDim.x * 256;
  for (long long i = blockIdx.x * 256LL + threadIdx.x; i < n; i += stride) {
    int nn = (int)(i % HWS); long long r = i / HWS;
    int o = (int)(r % NCLS); int b = (int)(r / NCLS);
    out[i] = tmp[((size_t)b * NPAD + o) * HWS + nn] + bias[o];
  }
}

// channel softmax (19) per pixel -> bf16 predT [B,HW,64], predB [B,64,HW], zero pad
__global__ void pred_softmax_k(const float* __restrict__ aux2,
                               u16* __restrict__ predT, u16* __restrict__ predB) {
  long long n = (long long)BATCH * HWS;
  long long stride = (long long)gridDim.x * 256;
  for (long long i = blockIdx.x * 256LL + threadIdx.x; i < n; i += stride) {
    int nn = (int)(i % HWS); int b = (int)(i / HWS);
    const float* p = aux2 + (size_t)b * NCLS * HWS + nn;
    float vals[NCLS]; float mx = -3.4e38f;
    #pragma unroll
    for (int c = 0; c < NCLS; ++c) { vals[c] = p[(size_t)c * HWS]; mx = fmaxf(mx, vals[c]); }
    float s = 0.f;
    #pragma unroll
    for (int c = 0; c < NCLS; ++c) { vals[c] = __expf(vals[c] - mx); s += vals[c]; }
    float inv = 1.f / s;
    #pragma unroll
    for (int c = 0; c < NPAD; ++c) {
      u16 h = (c < NCLS) ? f2bf(vals[c] * inv) : (u16)0;
      predT[((size_t)b * HWS + nn) * NPAD + c] = h;
      predB[((size_t)b * NPAD + c) * HWS + nn] = h;
    }
  }
}

// per-row: rel = softmax(rel_row) * sigmoid(gate_row), in place
__global__ void __launch_bounds__(256)
softmax_gate_k(float* __restrict__ rel, const float* __restrict__ gate) {
  __shared__ float sh[256];
  long long row = blockIdx.x;                // b*HW + n
  float* L = rel + row * (long long)HWS;
  const float* G = gate + row * (long long)HWS;
  int t = threadIdx.x;
  float mx = -3.4e38f;
  for (int m = t; m < HWS; m += 256) mx = fmaxf(mx, L[m]);
  sh[t] = mx; __syncthreads();
  for (int o = 128; o > 0; o >>= 1) { if (t < o) sh[t] = fmaxf(sh[t], sh[t + o]); __syncthreads(); }
  mx = sh[0]; __syncthreads();
  float s = 0.f;
  for (int m = t; m < HWS; m += 256) s += __expf(L[m] - mx);
  sh[t] = s; __syncthreads();
  for (int o = 128; o > 0; o >>= 1) { if (t < o) sh[t] += sh[t + o]; __syncthreads(); }
  float inv = 1.f / sh[0]; __syncthreads();
  for (int m = t; m < HWS; m += 256)
    L[m] = __expf(L[m] - mx) * inv / (1.f + __expf(-G[m]));
}

// aug_relation[b,n,q] = sum_p rel[b,n,shift_p(q)] * lr[b,q,p]  -> bf16
__global__ void augrel_k(const float* __restrict__ rel, const float* __restrict__ lr,
                         u16* __restrict__ out) {
  long long n = (long long)BATCH * HWS * HWS;
  long long stride = (long long)gridDim.x * 256;
  for (long long i = blockIdx.x * 256LL + threadIdx.x; i < n; i += stride) {
    int qq = (int)(i % HWS); long long r = i / HWS;
    int nn = (int)(r % HWS); int b = (int)(r / HWS);
    int h = qq / HS, w = qq % HS;
    const float* R = rel + ((size_t)b * HWS + nn) * HWS;
    const float* L = lr + ((size_t)b * HWS + qq) * 9;
    float s = 0.f;
    #pragma unroll
    for (int p = 0; p < 9; ++p) {
      int hh = h + p / 3 - 1, ww = w + p % 3 - 1;
      if (hh >= 0 && hh < HS && ww >= 0 && ww < HS) s += R[hh * HS + ww] * L[p];
    }
    out[i] = f2bf(s);
  }
}

// out_pre[b,c,n] = gamma * aug[b,n,c] + x[b,c,n]
__global__ void outpre_k(const float* __restrict__ aug, const float* __restrict__ x,
                         const float* __restrict__ gamma, float* __restrict__ out) {
  long long n = (long long)BATCH * CH * HWS;
  long long stride = (long long)gridDim.x * 256;
  float gm = gamma[0];
  for (long long i = blockIdx.x * 256LL + threadIdx.x; i < n; i += stride) {
    int nn = (int)(i % HWS); long long r = i / HWS;
    int c = (int)(r % CH); int b = (int)(r / CH);
    out[i] = gm * aug[((size_t)b * HWS + nn) * CH + c] + x[i];
  }
}

// align_corners bilinear 48->96, write bf16 [B,C,96,96]
__global__ void upsample_bf16_k(const float* __restrict__ src, u16* __restrict__ dst) {
  long long n = (long long)BATCH * CH * HW2;
  long long stride = (long long)gridDim.x * 256;
  const float sc = 47.f / 95.f;
  for (long long i = blockIdx.x * 256LL + threadIdx.x; i < n; i += stride) {
    int ow = (int)(i % H2); long long r = i / H2;
    int oh = (int)(r % H2); long long bc = r / H2;
    float fh = oh * sc, fw = ow * sc;
    int h0 = (int)fh, w0 = (int)fw;
    int h1 = h0 + 1 > 47 ? 47 : h0 + 1;
    int w1 = w0 + 1 > 47 ? 47 : w0 + 1;
    float ah = fh - h0, aw = fw - w0;
    const float* p = src + bc * (long long)HWS;
    float v = (1.f - ah) * ((1.f - aw) * p[h0 * HS + w0] + aw * p[h0 * HS + w1]) +
              ah * ((1.f - aw) * p[h1 * HS + w0] + aw * p[h1 * HS + w1]);
    dst[i] = f2bf(v);
  }
}

// ---------------------------------------------------------------------------
// Orchestration
// ---------------------------------------------------------------------------
extern "C" void kernel_launch(void* const* d_in, const int* in_sizes, int n_in,
                              void* d_out, int out_size, void* d_ws, size_t ws_size,
                              hipStream_t stream) {
  (void)in_sizes; (void)n_in; (void)out_size; (void)ws_size;
  const float* x    = (const float*)d_in[0];
  const float* x2   = (const float*)d_in[1];
  const float* w1   = (const float*)d_in[2];
  const float* w2   = (const float*)d_in[3];
  const float* w3   = (const float*)d_in[4];
  const float* gam  = (const float*)d_in[5];
  const float* wo1  = (const float*)d_in[6];
  const float* bn1g = (const float*)d_in[7];
  const float* bn1b = (const float*)d_in[8];
  const float* wo2  = (const float*)d_in[9];
  const float* bo2  = (const float*)d_in[10];
  const float* wf   = (const float*)d_in[11];
  const float* bnfg = (const float*)d_in[12];
  const float* bnfb = (const float*)d_in[13];

  float* out_main = (float*)d_out;                                   // [2,512,96,96]
  float* aux2_out = (float*)d_out + (size_t)BATCH * CH * HW2;        // [2,19,48,48]

  char* wsp = (char*)d_ws;
  size_t off = 0;
  auto carve = [&](size_t bytes) -> void* {
    off = (off + 255) & ~(size_t)255;
    void* p = wsp + off; off += bytes; return p;
  };
  const size_t BCHW  = (size_t)BATCH * CH * HWS;     // 2.36M
  const size_t BQK   = (size_t)BATCH * C8 * HWS;     // 294K
  const size_t BREL  = (size_t)BATCH * HWS * HWS;    // 10.6M

  u16*  xbf    = (u16*)carve(BCHW * 2);
  u16*  x2bf   = (u16*)carve((size_t)BATCH * 1024 * HWS * 2);
  u16*  w1bf   = (u16*)carve((size_t)C8 * CH * 2);
  u16*  w2bf   = (u16*)carve((size_t)C8 * CH * 2);
  u16*  w3bf   = (u16*)carve((size_t)CH * CH * 2);
  u16*  wo1bf  = (u16*)carve((size_t)CH * 1024 * 9 * 2);
  u16*  wo2bf  = (u16*)carve((size_t)NPAD * CH * 2);
  u16*  wfbf   = (u16*)carve((size_t)CH * CH * 9 * 2);
  float* qf    = (float*)carve(BQK * 4);
  float* kf    = (float*)carve(BQK * 4);
  float* vf    = (float*)carve(BCHW * 4);
  u16*  kbf    = (u16*)carve(BQK * 2);
  u16*  qTbf   = (u16*)carve(BQK * 2);
  u16*  vTbf   = (u16*)carve(BCHW * 2);
  float* lr    = (float*)carve((size_t)BATCH * HWS * 9 * 4);
  float* c1out = (float*)carve(BCHW * 4);
  float* mean1 = (float*)carve(CH * 4);
  float* var1  = (float*)carve(CH * 4);
  u16*  hbf    = (u16*)carve(BCHW * 2);
  float* a2tmp = (float*)carve((size_t)BATCH * NPAD * HWS * 4);
  u16*  predT  = (u16*)carve((size_t)BATCH * HWS * NPAD * 2);
  u16*  predB  = (u16*)carve((size_t)BATCH * NPAD * HWS * 2);
  float* rel   = (float*)carve(BREL * 4);     // 42.5 MB
  float* gate  = (float*)carve(BREL * 4);     // 42.5 MB
  u16*  augrel = (u16*)carve(BREL * 2);       // 21.2 MB
  float* aug   = (float*)carve(BCHW * 4);
  float* opre  = (float*)carve(BCHW * 4);
  float* meanf = (float*)carve(CH * 4);
  float* varf  = (float*)carve(CH * 4);
  // dead-buffer reuse: rel dies after augrel_k, gate dies after softmax_gate_k
  u16*  upbf    = (u16*)rel;    // needs 18.9 MB <= 42.5 MB
  float* cfout  = gate;         // needs 37.7 MB <= 42.5 MB

  auto nblk = [](long long n) -> unsigned {
    long long b = (n + 255) / 256;
    if (b < 1) b = 1; if (b > 262144) b = 262144;
    return (unsigned)b;
  };

  // --- precision converts -------------------------------------------------
  cvt_bf16_k<<<nblk(BCHW), 256, 0, stream>>>(x, xbf, (long long)BCHW);
  cvt_bf16_k<<<nblk((long long)BATCH*1024*HWS), 256, 0, stream>>>(x2, x2bf, (long long)BATCH*1024*HWS);
  cvt_bf16_k<<<nblk(C8*CH), 256, 0, stream>>>(w1, w1bf, (long long)C8*CH);
  cvt_bf16_k<<<nblk(C8*CH), 256, 0, stream>>>(w2, w2bf, (long long)C8*CH);
  cvt_bf16_k<<<nblk(CH*CH), 256, 0, stream>>>(w3, w3bf, (long long)CH*CH);
  cvt_bf16_k<<<nblk((long long)CH*1024*9), 256, 0, stream>>>(wo1, wo1bf, (long long)CH*1024*9);
  cvt_pad_rows_k<<<nblk(NPAD*CH), 256, 0, stream>>>(wo2, wo2bf, NCLS, NPAD, CH);
  cvt_bf16_k<<<nblk((long long)CH*CH*9), 256, 0, stream>>>(wf, wfbf, (long long)CH*CH*9);

  // --- q / k / v projections (WMMA) --------------------------------------
  gemm_bf16_k<<<dim3(HWS/BLK_N, C8/BLK_M, BATCH), 256, 0, stream>>>(
      w1bf, xbf, qf, C8, HWS, CH, CH, HWS, HWS, 0, (long long)CH*HWS, (long long)C8*HWS);
  gemm_bf16_k<<<dim3(HWS/BLK_N, C8/BLK_M, BATCH), 256, 0, stream>>>(
      w2bf, xbf, kf, C8, HWS, CH, CH, HWS, HWS, 0, (long long)CH*HWS, (long long)C8*HWS);
  gemm_bf16_k<<<dim3(HWS/BLK_N, CH/BLK_M, BATCH), 256, 0, stream>>>(
      w3bf, xbf, vf, CH, HWS, CH, CH, HWS, HWS, 0, (long long)CH*HWS, (long long)CH*HWS);

  // layout converts for attention GEMMs
  cvt_bf16_k<<<nblk(BQK), 256, 0, stream>>>(kf, kbf, (long long)BQK);
  tcvt_bf16_k<<<dim3(nblk(C8*HWS),1,BATCH), 256, 0, stream>>>(qf, qTbf, C8, HWS,
      (long long)C8*HWS, (long long)C8*HWS);
  tcvt_bf16_k<<<dim3(nblk(CH*HWS),1,BATCH), 256, 0, stream>>>(vf, vTbf, CH, HWS,
      (long long)CH*HWS, (long long)CH*HWS);

  // local relation (small, f32 VALU)
  local_rel_k<<<nblk((long long)BATCH*HWS*9), 256, 0, stream>>>(qf, kf, lr);

  // --- aux head: conv3x3 1024->512 @48x48 (implicit GEMM, WMMA) -----------
  conv3x3_gemm_k<<<dim3((BATCH*HWS)/BLK_N, CH/BLK_M), 256, 0, stream>>>(
      wo1bf, x2bf, c1out, CH, 1024, HS, HS, BATCH);
  bn_stats_k<<<CH, 256, 0, stream>>>(c1out, mean1, var1, CH, HWS, BATCH);
  bn_relu_bf16_k<<<nblk(BCHW), 256, 0, stream>>>(c1out, mean1, var1, bn1g, bn1b,
      hbf, CH, HWS, (long long)BCHW);
  gemm_bf16_k<<<dim3(HWS/BLK_N, 1, BATCH), 256, 0, stream>>>(
      wo2bf, hbf, a2tmp, NPAD, HWS, CH, CH, HWS, HWS, 0, (long long)CH*HWS,
      (long long)NPAD*HWS);
  aux2_fin_k<<<nblk((long long)BATCH*NCLS*HWS), 256, 0, stream>>>(a2tmp, bo2, aux2_out);

  // prediction-gated mask: class softmax, gate = predT*pred (WMMA, K=64 pad)
  pred_softmax_k<<<nblk((long long)BATCH*HWS), 256, 0, stream>>>(aux2_out, predT, predB);
  gemm_bf16_k<<<dim3(HWS/BLK_N, HWS/BLK_M, BATCH), 256, 0, stream>>>(
      predT, predB, gate, HWS, HWS, NPAD, NPAD, HWS, HWS,
      (long long)HWS*NPAD, (long long)NPAD*HWS, (long long)HWS*HWS);

  // dense attention logits qT*k (WMMA, K=64), then fused softmax*sigmoid(gate)
  gemm_bf16_k<<<dim3(HWS/BLK_N, HWS/BLK_M, BATCH), 256, 0, stream>>>(
      qTbf, kbf, rel, HWS, HWS, C8, C8, HWS, HWS,
      (long long)HWS*C8, (long long)C8*HWS, (long long)HWS*HWS);
  softmax_gate_k<<<BATCH*HWS, 256, 0, stream>>>(rel, gate);

  // unfold-over-keys stencil -> bf16 A operand, then aug = augrel * vT (WMMA)
  augrel_k<<<nblk((long long)BREL), 256, 0, stream>>>(rel, lr, augrel);
  gemm_bf16_k<<<dim3(CH/BLK_N, HWS/BLK_M, BATCH), 256, 0, stream>>>(
      augrel, vTbf, aug, HWS, CH, HWS, HWS, CH, CH,
      (long long)HWS*HWS, (long long)HWS*CH, (long long)HWS*CH);

  // residual + upsample (align_corners) + final conv3x3 512->512 @96x96
  outpre_k<<<nblk(BCHW), 256, 0, stream>>>(aug, x, gam, opre);
  upsample_bf16_k<<<nblk((long long)BATCH*CH*HW2), 256, 0, stream>>>(opre, upbf);
  conv3x3_gemm_k<<<dim3((BATCH*HW2)/BLK_N, CH/BLK_M), 256, 0, stream>>>(
      wfbf, upbf, cfout, CH, CH, H2, H2, BATCH);
  bn_stats_k<<<CH, 256, 0, stream>>>(cfout, meanf, varf, CH, HW2, BATCH);
  bn_relu_f32_k<<<nblk((long long)BATCH*CH*HW2), 256, 0, stream>>>(
      cfout, meanf, varf, bnfg, bnfb, out_main, CH, HW2, (long long)BATCH*CH*HW2);
}
